// PrefilledAttention_56478819942745
// MI455X (gfx1250) — matrix operand order repaired and checked by reading, not compiled, not run on hardware
//
#include <hip/hip_runtime.h>
#include <hip/hip_bf16.h>
#include <math.h>
#include <stdint.h>

typedef __attribute__((ext_vector_type(16))) __bf16 v16bf;
typedef __attribute__((ext_vector_type(8)))  __bf16 v8bf;
typedef __attribute__((ext_vector_type(4)))  __bf16 v4bf;
typedef __attribute__((ext_vector_type(8)))  float  v8f;
typedef __attribute__((ext_vector_type(4)))  unsigned int v4u;
typedef __attribute__((ext_vector_type(8)))  int v8i;
typedef __attribute__((ext_vector_type(4)))  int v4i;

#define LDSE 2064  // padded LDS row stride in bf16 elems (2048 keys + 16 pad)

#ifndef __has_builtin
#define __has_builtin(x) 0
#endif
#if defined(__HIP_DEVICE_COMPILE__) && __has_builtin(__builtin_amdgcn_tensor_load_to_lds)
#define LRA_HAS_TDM 1
#else
#define LRA_HAS_TDM 0
#endif

static __device__ __forceinline__ v16bf cat16(v8bf lo, v8bf hi) {
  v16bf r;
#pragma unroll
  for (int i = 0; i < 8; ++i) { r[i] = lo[i]; r[i + 8] = hi[i]; }
  return r;
}

#if LRA_HAS_TDM
// Issue one TDM load of a [16 rows x 128 cols] bf16 tile (row stride 128)
// from global into LDS at byte offset lds_off. D# packing per ISA 08 §8.
static __device__ __forceinline__ void tdm_load_q_tile(const __bf16* gsrc,
                                                       unsigned lds_off) {
  unsigned long long ga = (unsigned long long)(uintptr_t)gsrc;
  v4u g0;
  g0[0] = 1u;                                   // count=1, user mode
  g0[1] = lds_off;                              // lds_addr (bytes)
  g0[2] = (unsigned)(ga & 0xFFFFFFFFu);         // global_addr[31:0]
  g0[3] = (unsigned)((ga >> 32) & 0x1FFFFFFu)   // global_addr[56:32]
        | (2u << 30);                           // type=2 ("image")
  v8i g1;
  g1[0] = (int)(1u << 16);       // wg_mask=0, data_size=1 (2 bytes)
  g1[1] = (int)(128u << 16);     // tensor_dim0 = 128 (bits 79:48 low half)
  g1[2] = (int)(16384u << 16);   // tensor_dim1 = 16384 rows (bits 111:80 low half)
  g1[3] = (int)(128u << 16);     // tile_dim0 = 128 (bits 127:112)
  g1[4] = 16;                    // tile_dim1 = 16 rows
  g1[5] = 128;                   // tensor_dim0_stride = 128 elems
  g1[6] = 0;
  g1[7] = 0;
  v4i z4 = {};
#if __clang_major__ >= 23
  v8i z8 = {};
  __builtin_amdgcn_tensor_load_to_lds(g0, g1, z4, z4, z8, 0);
#else
  __builtin_amdgcn_tensor_load_to_lds(g0, g1, z4, z4, 0);
#endif
}
#endif

// ---------------------------------------------------------------- convert
__global__ __launch_bounds__(256) void lra_cvt_kernel(
    const float* __restrict__ in, __bf16* __restrict__ out, int n, float scale) {
  int i = (blockIdx.x * 256 + threadIdx.x) * 4;
  if (i + 3 < n) {
    float4 v = *(const float4*)(in + i);
    v4bf o;
    o[0] = (__bf16)(v.x * scale);
    o[1] = (__bf16)(v.y * scale);
    o[2] = (__bf16)(v.z * scale);
    o[3] = (__bf16)(v.w * scale);
    *(v4bf*)(out + i) = o;
  }
}

// ---------------------------------------------------------------- projection
// out = X[16384 x 1024] @ W^T  (W is [numN x 1024] bf16, row-major)
// transposed==0: out row-major [token][numN]   (q, k)
// transposed==1: out [batch][numN][2048]       (v^T), written with b128 stores
__global__ __launch_bounds__(256) void lra_proj_kernel(
    const __bf16* __restrict__ X, const __bf16* __restrict__ W,
    __bf16* __restrict__ out, int numN, int transposed) {
  const int tilesN = numN >> 4;
  const int lane = threadIdx.x & 31;
  const int nl = lane & 15;
  const int kh = lane >> 4;
  const int id = blockIdx.x * 8 + (threadIdx.x >> 5);
  const int t0 = (id / tilesN) * 16;  // token tile base (global token idx)
  const int n0 = (id % tilesN) * 16;  // feature tile base

  v8f acc = {};
  for (int kk = 0; kk < 1024; kk += 32) {
    // A fragment: 16x32 bf16 of X, lane row = t0+nl
    const __bf16* xp = X + (size_t)(t0 + nl) * 1024 + kk;
    v16bf a = cat16(*(const v8bf*)(xp + kh * 8),
                    *(const v8bf*)(xp + 16 + kh * 8));
    // B fragment: 32x16, B[kq][n] = W[n][kk+kq]; same k order as A
    const __bf16* wp = W + (size_t)(n0 + nl) * 1024 + kk;
    v16bf b = cat16(*(const v8bf*)(wp + kh * 8), *(const v8bf*)(wp + 16 + kh * 8));
    acc = __builtin_amdgcn_wmma_f32_16x16x32_bf16(false, a, false, b,
                                                  (short)0, acc, false, false);
  }

  if (transposed) {  // v^T: lane writes 8 consecutive tokens at fixed d -> one b128
    v8bf o;
#pragma unroll
    for (int i = 0; i < 8; ++i) o[i] = (__bf16)acc[i];
    const size_t base = (size_t)(t0 >> 11) * (size_t)numN * 2048 +
                        (size_t)(n0 + nl) * 2048 + (t0 & 2047) + 8 * kh;
    *(v8bf*)(out + base) = o;
  } else {           // row-major q/k: 8 scattered b16 stores (column of the tile)
#pragma unroll
    for (int i = 0; i < 8; ++i)
      out[(size_t)(t0 + i + 8 * kh) * numN + n0 + nl] = (__bf16)acc[i];
  }
}

// ---------------------------------------------------------------- attention
// One block = 16 query rows of one batch. 8 waves (wave32).
__global__ __launch_bounds__(256) void lra_attn_kernel(
    const __bf16* __restrict__ Q, const __bf16* __restrict__ K,
    const __bf16* __restrict__ VT, float* __restrict__ out) {
  extern __shared__ __align__(16) char smem[];
  __bf16* P   = (__bf16*)smem;                          // [16][LDSE]
  float* sums = (float*)(smem + 16 * LDSE * 2);         // [16]
  __bf16* Qs  = (__bf16*)(smem + 16 * LDSE * 2 + 64);   // [16][128] staged Q

  const int b    = blockIdx.y;
  const int q0   = blockIdx.x * 16;  // query tile within batch
  const int tid  = threadIdx.x;
  const int wave = tid >> 5;
  const int lane = tid & 31;
  const int nl   = lane & 15;
  const int kh   = lane >> 4;
  const size_t bTok = (size_t)b * 2048;

  // ---- Stage 0: stage the 16x128 Q tile into LDS --------------------------
#if LRA_HAS_TDM
  if (tid < 32) {                      // wave 0 drives the Tensor Data Mover
    tdm_load_q_tile(Q + (bTok + q0) * 128, (unsigned)(16 * LDSE * 2 + 64));
#if __has_builtin(__builtin_amdgcn_s_wait_tensorcnt)
    __builtin_amdgcn_s_wait_tensorcnt(0);
#else
    asm volatile("s_wait_tensorcnt 0x0" ::: "memory");
#endif
  }
#else
  ((v8bf*)Qs)[tid] = ((const v8bf*)(Q + (bTok + q0) * 128))[tid];
#endif
  __syncthreads();

  // ---- Stage 1: S = (Q*scale) K^T for 16 queries x 2048 keys --------------
  v16bf qa[4];
#pragma unroll
  for (int r = 0; r < 4; ++r) {
    const __bf16* p = Qs + nl * 128 + r * 32;
    qa[r] = cat16(*(const v8bf*)(p + kh * 8), *(const v8bf*)(p + 16 + kh * 8));
  }
  for (int sub = 0; sub < 16; ++sub) {
    const int t0 = wave * 256 + sub * 16;  // this wave's key subtile
    v8f acc = {};
#pragma unroll
    for (int r = 0; r < 4; ++r) {
      // B[kq][t] = K[t][r]; same k order as A at fixed key token
      const __bf16* p = K + (bTok + t0 + nl) * 128 + r * 32;
      v16bf bb = cat16(*(const v8bf*)(p + kh * 8), *(const v8bf*)(p + 16 + kh * 8));
      acc = __builtin_amdgcn_wmma_f32_16x16x32_bf16(false, qa[r], false, bb,
                                                    (short)0, acc, false, false);
    }
#pragma unroll
    for (int i = 0; i < 8; ++i)
      P[(i + 8 * kh) * LDSE + t0 + nl] = (__bf16)acc[i];
  }
  __syncthreads();

  // ---- Stage 2: row softmax (unnormalized exp; defer 1/sum to epilogue) ---
  {
    const int row = tid >> 4;  // 16 threads per row, two rows per wave
    const int c0  = tid & 15;
    float m = -3.0e38f;
#pragma unroll 8
    for (int c = c0; c < 2048; c += 16)
      m = fmaxf(m, (float)P[row * LDSE + c]);
#pragma unroll
    for (int off = 8; off > 0; off >>= 1)
      m = fmaxf(m, __shfl_xor(m, off, 16));
    float s = 0.f;
#pragma unroll 8
    for (int c = c0; c < 2048; c += 16) {
      float e = __expf((float)P[row * LDSE + c] - m);
      P[row * LDSE + c] = (__bf16)e;
      s += e;
    }
#pragma unroll
    for (int off = 8; off > 0; off >>= 1)
      s += __shfl_xor(s, off, 16);
    if (c0 == 0) sums[row] = s;
  }
  __syncthreads();

  // ---- Stage 3: O = exp(S) @ V; 1024 output cols split across 8 waves -----
  const int d0 = wave * 128;
  const __bf16* vbase = VT + ((size_t)b * 1024 + d0 + nl) * 2048;
  v8f acc[8];
#pragma unroll
  for (int f = 0; f < 8; ++f) acc[f] = (v8f){};
  for (int t0 = 0; t0 < 2048; t0 += 32) {
    // pull the VT stream 8 K-tiles (512B) ahead -> global_prefetch
    if (t0 + 256 < 2048) __builtin_prefetch(vbase + t0 + 256, 0, 3);
    // A fragment: P rows = queries, K = key index (LDS, b128 loads)
    const __bf16* pp = P + nl * LDSE + t0;
    v16bf pa = cat16(*(const v8bf*)(pp + kh * 8),
                     *(const v8bf*)(pp + 16 + kh * 8));
#pragma unroll
    for (int f = 0; f < 8; ++f) {
      // B[kq][d] = V[t0+kq][d] = VT[b][d][t0+kq]; contiguous along t
      const __bf16* vp = vbase + (size_t)f * 16 * 2048 + t0;
      v16bf bv = cat16(*(const v8bf*)(vp + kh * 8), *(const v8bf*)(vp + 16 + kh * 8));
      acc[f] = __builtin_amdgcn_wmma_f32_16x16x32_bf16(false, pa, false, bv,
                                                       (short)0, acc[f], false, false);
    }
  }

  // ---- Stage 4: normalize by row sum (8 reciprocals, then FMA) ------------
  float inv[8];
#pragma unroll
  for (int i = 0; i < 8; ++i) inv[i] = 1.0f / sums[i + 8 * kh];
#pragma unroll
  for (int f = 0; f < 8; ++f) {
#pragma unroll
    for (int i = 0; i < 8; ++i) {
      const int m = i + 8 * kh;
      const size_t o = (bTok + q0 + m) * 1024 + d0 + f * 16 + nl;
      out[o] = acc[f][i] * inv[i];
    }
  }
}

// ---------------------------------------------------------------- launcher
extern "C" void kernel_launch(void* const* d_in, const int* in_sizes, int n_in,
                              void* d_out, int out_size, void* d_ws, size_t ws_size,
                              hipStream_t stream) {
  const float* x  = (const float*)d_in[0];   // [8,2048,1024]
  const float* Wq = (const float*)d_in[1];   // [128,1024]
  const float* Wk = (const float*)d_in[2];   // [128,1024]
  const float* Wv = (const float*)d_in[3];   // [1024,1024]
  float* out = (float*)d_out;                // [8,2048,1024] fp32

  const size_t NX = 8u * 2048u * 1024u;      // 16,777,216
  const size_t NW = 128u * 1024u;            // 131,072
  const size_t NV = 1024u * 1024u;           // 1,048,576
  const size_t NQK = 16384u * 128u;          // 2,097,152

  char* p = (char*)d_ws;
  __bf16* xb  = (__bf16*)p; p += NX * 2;
  __bf16* wqb = (__bf16*)p; p += NW * 2;
  __bf16* wkb = (__bf16*)p; p += NW * 2;
  __bf16* wvb = (__bf16*)p; p += NV * 2;
  __bf16* qb  = (__bf16*)p; p += NQK * 2;
  __bf16* kb  = (__bf16*)p; p += NQK * 2;
  __bf16* vT  = (__bf16*)p;  // [8][1024][2048] bf16

  const float scl = 0.08838834764831845f;  // 1/sqrt(128), folded into Wq

  lra_cvt_kernel<<<NX / 1024, 256, 0, stream>>>(x, xb, (int)NX, 1.0f);
  lra_cvt_kernel<<<NW / 1024, 256, 0, stream>>>(Wq, wqb, (int)NW, scl);
  lra_cvt_kernel<<<NW / 1024, 256, 0, stream>>>(Wk, wkb, (int)NW, 1.0f);
  lra_cvt_kernel<<<NV / 1024, 256, 0, stream>>>(Wv, wvb, (int)NV, 1.0f);

  // q: 1024x8 tiles, k: 1024x8 tiles, v: 1024x64 tiles; 8 tiles per block
  lra_proj_kernel<<<1024, 256, 0, stream>>>(xb, wqb, qb, 128, 0);
  lra_proj_kernel<<<1024, 256, 0, stream>>>(xb, wkb, kb, 128, 0);
  lra_proj_kernel<<<8192, 256, 0, stream>>>(xb, wvb, vT, 1024, 1);

  const size_t shmem = (size_t)16 * LDSE * 2 + 64 + 4096;  // ~70 KB of 320 KB
  (void)hipFuncSetAttribute((const void*)lra_attn_kernel,
                            hipFuncAttributeMaxDynamicSharedMemorySize, (int)shmem);
  lra_attn_kernel<<<dim3(128, 8), 256, shmem, stream>>>(qb, kb, vT, out);
}
